// _PAMBlock_1726576855063
// MI455X (gfx1250) — matrix operand order repaired and checked by reading, not compiled, not run on hardware
//
#include <hip/hip_runtime.h>
#include <hip/hip_bf16.h>

// ---------------------------------------------------------------------------
// PAM block attention for MI455X (gfx1250, wave32, WMMA 16x16x32 f16).
// Pipeline: prep -> WMMA conv1x1 -> BN stats -> pack f16 layouts -> flash attn
// with double-buffered async (GLOBAL_LOAD_ASYNC_TO_LDS_B128) K/V staging.
// ---------------------------------------------------------------------------

typedef __attribute__((ext_vector_type(16))) _Float16 v16h;
typedef __attribute__((ext_vector_type(8)))  _Float16 v8h;
typedef __attribute__((ext_vector_type(8)))  float    v8f;
typedef __attribute__((ext_vector_type(4)))  int      v4i;

typedef __attribute__((address_space(1))) v4i gv4i;   // global int4
typedef __attribute__((address_space(3))) v4i lv4i;   // LDS int4

#define WMMA_F16(a, b, c) \
  __builtin_amdgcn_wmma_f32_16x16x32_f16(false, (a), false, (b), (short)0, (c), false, false)

#if defined(__has_builtin)
# if __has_builtin(__builtin_amdgcn_global_load_async_to_lds_b128)
#  define HAVE_ASYNC 1
# endif
# if __has_builtin(__builtin_amdgcn_s_wait_asynccnt)
#  define HAVE_WAIT_ASYNC 1
# endif
#endif

static __device__ __forceinline__ v16h mk16(v8h lo, v8h hi) {
  v16h r;
#pragma unroll
  for (int i = 0; i < 8; ++i) { r[i] = lo[i]; r[i + 8] = hi[i]; }
  return r;
}

// 16-byte global -> LDS copy: async path when available, else through VGPRs
static __device__ __forceinline__ void copy16_async(const _Float16* g, _Float16* l) {
#if defined(HAVE_ASYNC)
  __builtin_amdgcn_global_load_async_to_lds_b128((gv4i*)g, (lv4i*)l, 0, 0);
#else
  *(v8h*)l = *(const v8h*)g;
#endif
}

static __device__ __forceinline__ void stage_fence() {
#if defined(HAVE_ASYNC)
# if defined(HAVE_WAIT_ASYNC)
  __builtin_amdgcn_s_wait_asynccnt(0);
# else
  asm volatile("s_wait_asynccnt 0x0" ::: "memory");
# endif
#endif
  __syncthreads();
}

// Problem constants
// x: (N=4, C=64, H=64, W2=128); S=2 -> blocks 32x32, L = 32*32*2 = 2048
// pairs = N*S*S = 16; q/k: CK=32 ch; v: CV=64 ch; P = N*H*W2 = 32768 pixels
#define L_TOK 2048
#define NPIX  32768

// ---------------------------------------------------------------------------
// Prep: weights (o-major f16), biases, x transposed to (pixel, c) f16
// ---------------------------------------------------------------------------
__global__ void prep_wb(const float* __restrict__ Wq, const float* __restrict__ bq,
                        const float* __restrict__ Wk, const float* __restrict__ bk,
                        const float* __restrict__ Wv, const float* __restrict__ bv,
                        _Float16* __restrict__ Wall, float* __restrict__ ball) {
  int o = threadIdx.x;   // 0..127
  const float* src; float bias;
  if (o < 32)       { src = Wq + o * 64;        bias = bq[o]; }
  else if (o < 64)  { src = Wk + (o - 32) * 64; bias = bk[o - 32]; }
  else              { src = Wv + (o - 64) * 64; bias = bv[o - 64]; }
  ball[o] = bias;
  for (int c = 0; c < 64; ++c) Wall[o * 64 + c] = (_Float16)src[c];
}

__global__ void prep_xt(const float* __restrict__ x, _Float16* __restrict__ xt) {
  int idx = blockIdx.x * 256 + threadIdx.x;       // < NPIX*64
  int p = idx >> 6, c = idx & 63;
  int n = p >> 13, h = (p >> 7) & 63, w = p & 127;
  xt[idx] = (_Float16)x[(((size_t)n * 64 + c) * 64 + h) * 128 + w];
}

// ---------------------------------------------------------------------------
// Conv1x1 via WMMA: per wave, 16-pixel x 128-outch tile, K=64 in 2 steps.
// praw: (pixel, 128) f32 raw projections (q:0-31, k:32-63, v:64-127)
// ---------------------------------------------------------------------------
__global__ void proj_kernel(const _Float16* __restrict__ xt,
                            const _Float16* __restrict__ Wall,
                            const float* __restrict__ ball,
                            float* __restrict__ praw) {
  __shared__ __align__(16) _Float16 wlds[128 * 64];
  int tid = threadIdx.x;
  for (int i = tid; i < 128 * 64; i += 256) wlds[i] = Wall[i];
  __syncthreads();

  int wave = tid >> 5, lane = tid & 31, lm = lane & 15, lh = lane >> 4;
  int pbase = (blockIdx.x * 8 + wave) * 16;

  // A operand = X^T tile (16 pixels x 64 ch), two K=32 steps
  const _Float16* xr = xt + (size_t)(pbase + lm) * 64;
  v16h ax0 = mk16(*(const v8h*)(xr + 16 * lh), *(const v8h*)(xr + 8 + 16 * lh));
  v16h ax1 = mk16(*(const v8h*)(xr + 32 + 16 * lh), *(const v8h*)(xr + 40 + 16 * lh));

#pragma unroll
  for (int ot = 0; ot < 8; ++ot) {
    // B operand = W^T (c x o): lane -> o, half0 c0-15 / half1 c16-31
    const _Float16* wr = wlds + (ot * 16 + lm) * 64 + 16 * lh;
    v16h bw0 = mk16(*(const v8h*)(wr),      *(const v8h*)(wr + 8));
    v16h bw1 = mk16(*(const v8h*)(wr + 32), *(const v8h*)(wr + 40));
    v8f acc = {};
    acc = WMMA_F16(ax0, bw0, acc);
    acc = WMMA_F16(ax1, bw1, acc);
    float bias = ball[ot * 16 + lm];
#pragma unroll
    for (int r = 0; r < 8; ++r) {
      int pix = pbase + r + 8 * lh;                    // D: VGPR r, half -> M row
      praw[(size_t)pix * 128 + ot * 16 + lm] = acc[r] + bias;
    }
  }
}

// ---------------------------------------------------------------------------
// BN stats per channel (deterministic block reduction), channels 0..63 (q,k)
// ---------------------------------------------------------------------------
__global__ void stats_kernel(const float* __restrict__ praw,
                             const float* __restrict__ gq, const float* __restrict__ betaq,
                             const float* __restrict__ gk, const float* __restrict__ betak,
                             float* __restrict__ affA, float* __restrict__ affC) {
  __shared__ float s1[256], s2[256];
  int o = blockIdx.x, tid = threadIdx.x;
  float a = 0.f, b = 0.f;
  for (int p = tid; p < NPIX; p += 256) {
    float v = praw[(size_t)p * 128 + o];
    a += v; b += v * v;
  }
  s1[tid] = a; s2[tid] = b; __syncthreads();
  for (int st = 128; st > 0; st >>= 1) {
    if (tid < st) { s1[tid] += s1[tid + st]; s2[tid] += s2[tid + st]; }
    __syncthreads();
  }
  if (tid == 0) {
    float mean = s1[0] / (float)NPIX;
    float var  = s2[0] / (float)NPIX - mean * mean;
    float g  = (o < 32) ? gq[o]    : gk[o - 32];
    float bb = (o < 32) ? betaq[o] : betak[o - 32];
    float aa = g * rsqrtf(var + 1e-5f);
    affA[o] = aa;
    affC[o] = bb - mean * aa;
  }
}

// ---------------------------------------------------------------------------
// Pack into attention layouts:
//   qbt/kbt: (pair, l, ck) f16   (pair = n*4 + (i*2+j))
//   vbt:     (pair, cv, l) f16
//   l = (h_local*32 + w_local)*2 + side ; h = i*32+h_local ; w2 = side*64 + j*32 + w_local
// ---------------------------------------------------------------------------
__global__ void pack_kernel(const float* __restrict__ praw,
                            const float* __restrict__ affA, const float* __restrict__ affC,
                            _Float16* __restrict__ qbt, _Float16* __restrict__ kbt,
                            _Float16* __restrict__ vbt) {
  int idx = blockIdx.x * blockDim.x + threadIdx.x;   // 0..32767 = pair*2048 + l
  int pair = idx >> 11, l = idx & 2047;
  int n = pair >> 2, b = pair & 3, bi = b >> 1, bj = b & 1;
  int side = l & 1, wl = (l >> 1) & 31, hl = l >> 6;
  int h = bi * 32 + hl, w = side * 64 + bj * 32 + wl;
  size_t p = (size_t)(n * 64 + h) * 128 + w;         // pixel index (n,h,w2)
  const float* row = praw + p * 128;

  _Float16* q = qbt + (size_t)idx * 32;
  _Float16* k = kbt + (size_t)idx * 32;
#pragma unroll
  for (int o = 0; o < 32; ++o) {
    q[o] = (_Float16)(row[o] * affA[o] + affC[o]);
    k[o] = (_Float16)(row[32 + o] * affA[32 + o] + affC[32 + o]);
  }
  _Float16* vrow = vbt + (size_t)pair * 64 * L_TOK + l;
#pragma unroll
  for (int cv = 0; cv < 64; ++cv) vrow[(size_t)cv * L_TOK] = (_Float16)row[64 + cv];
}

// ---------------------------------------------------------------------------
// Flash attention per (pair, 128-row block): 8 waves x 16 query rows.
// Workgroup stages K (32x32) + V (64x32) chunks into double-buffered LDS via
// async global->LDS copies; inner loop: 2 WMMA (S) + online softmax + 4 WMMA
// (P·V^T) with P transposed through a per-wave LDS staging tile.
// ---------------------------------------------------------------------------
__global__ void attn_kernel(const _Float16* __restrict__ qbt,
                            const _Float16* __restrict__ kbt,
                            const _Float16* __restrict__ vbt,
                            float* __restrict__ out) {
  __shared__ __align__(16) _Float16 klds[2][32][32];   // [buf][m_local][ck]
  __shared__ __align__(16) _Float16 vlds[2][64][32];   // [buf][cv][m_local]
  __shared__ __align__(16) _Float16 plds[8][16][32];   // per-wave P staging
  int tid = threadIdx.x, wave = tid >> 5, lane = tid & 31;
  int lm = lane & 15, lh = lane >> 4;
  int pair = blockIdx.x >> 4, rowblk = blockIdx.x & 15;
  int lbase = rowblk * 128 + wave * 16;

  const _Float16* qb = qbt + (size_t)pair * L_TOK * 32;
  const _Float16* kb = kbt + (size_t)pair * L_TOK * 32;
  const _Float16* vb = vbt + (size_t)pair * 64 * L_TOK;

  // cooperative stage of one 32-column chunk (K: 128x16B, V: 256x16B pieces)
#define STAGE(bufi, mbase)                                                   \
  do {                                                                       \
    if (tid < 128) {                                                         \
      int m_ = tid >> 2, pt_ = tid & 3;                                      \
      copy16_async(kb + (size_t)((mbase) + m_) * 32 + pt_ * 8,               \
                   &klds[bufi][m_][pt_ * 8]);                                \
    }                                                                        \
    int cv_ = tid >> 2, pt2_ = tid & 3;                                      \
    copy16_async(vb + (size_t)cv_ * L_TOK + (mbase) + pt2_ * 8,              \
                 &vlds[bufi][cv_][pt2_ * 8]);                                \
  } while (0)

  // A operand = Q^T tile (16 l-rows x 32 ck), loaded once
  const _Float16* qrow = qb + (size_t)(lbase + lm) * 32;
  v16h aq = mk16(*(const v8h*)(qrow + 16 * lh), *(const v8h*)(qrow + 8 + 16 * lh));

  v8f acc0 = {}, acc1 = {}, acc2 = {}, acc3 = {};
  float rmax[8], rsum[8];
#pragma unroll
  for (int r = 0; r < 8; ++r) { rmax[r] = -1e30f; rsum[r] = 0.f; }
  const float sc = 0.17677669529663687f;   // 1/sqrt(CK=32)

  STAGE(0, 0);
  stage_fence();

  int buf = 0;
  for (int mb = 0; mb < L_TOK; mb += 32, buf ^= 1) {
    if (mb + 32 < L_TOK) STAGE(buf ^ 1, mb + 32);   // prefetch next chunk

    // B operands = K (ck x m) from LDS: lane -> m, half0 ck0-15 / half1 ck16-31
    const _Float16* kr0 = &klds[buf][lm][16 * lh];
    const _Float16* kr1 = &klds[buf][16 + lm][16 * lh];
    v16h bk0 = mk16(*(const v8h*)kr0, *(const v8h*)(kr0 + 8));
    v16h bk1 = mk16(*(const v8h*)kr1, *(const v8h*)(kr1 + 8));
    v8f z = {};
    v8f s0 = WMMA_F16(aq, bk0, z);    // rows l (VGPR), cols m 0..15 (lanes)
    v8f s1 = WMMA_F16(aq, bk1, z);    // cols m 16..31

    float p0[8], p1[8];
#pragma unroll
    for (int r = 0; r < 8; ++r) { p0[r] = s0[r] * sc; p1[r] = s1[r] * sc; }

#pragma unroll
    for (int r = 0; r < 8; ++r) {
      // row chunk-max across the 16 lanes of each half (xor < 16 stays in half)
      float m = fmaxf(p0[r], p1[r]);
      m = fmaxf(m, __shfl_xor(m, 1, 32));
      m = fmaxf(m, __shfl_xor(m, 2, 32));
      m = fmaxf(m, __shfl_xor(m, 4, 32));
      m = fmaxf(m, __shfl_xor(m, 8, 32));
      float nm = fmaxf(rmax[r], m);
      float alpha = __expf(rmax[r] - nm);
      rmax[r] = nm;
      p0[r] = __expf(p0[r] - nm);
      p1[r] = __expf(p1[r] - nm);
      float ls = p0[r] + p1[r];
      ls += __shfl_xor(ls, 1, 32);
      ls += __shfl_xor(ls, 2, 32);
      ls += __shfl_xor(ls, 4, 32);
      ls += __shfl_xor(ls, 8, 32);
      rsum[r] = rsum[r] * alpha + ls;
      acc0[r] *= alpha; acc1[r] *= alpha; acc2[r] *= alpha; acc3[r] *= alpha;
    }

    // stage P (16 l x 32 m) to per-wave LDS, reload in A-operand layout
#pragma unroll
    for (int r = 0; r < 8; ++r) {
      plds[wave][r + 8 * lh][lm]      = (_Float16)p0[r];
      plds[wave][r + 8 * lh][16 + lm] = (_Float16)p1[r];
    }
    // same-wave DS ops are in order; compiler inserts s_wait_dscnt for the RAW
    v16h ap = mk16(*(const v8h*)&plds[wave][lm][16 * lh],
                   *(const v8h*)&plds[wave][lm][8 + 16 * lh]);

    // B operands = V^T (m x cv) from LDS: lane -> cv, half0 m0-15 / half1 m16-31
    const _Float16* vr0 = &vlds[buf][lm][16 * lh];
    const _Float16* vr1 = &vlds[buf][16 + lm][16 * lh];
    const _Float16* vr2 = &vlds[buf][32 + lm][16 * lh];
    const _Float16* vr3 = &vlds[buf][48 + lm][16 * lh];
    acc0 = WMMA_F16(ap, mk16(*(const v8h*)vr0, *(const v8h*)(vr0 + 8)), acc0);
    acc1 = WMMA_F16(ap, mk16(*(const v8h*)vr1, *(const v8h*)(vr1 + 8)), acc1);
    acc2 = WMMA_F16(ap, mk16(*(const v8h*)vr2, *(const v8h*)(vr2 + 8)), acc2);
    acc3 = WMMA_F16(ap, mk16(*(const v8h*)vr3, *(const v8h*)(vr3 + 8)), acc3);

    // wait for prefetched chunk + all waves done with this buffer
    stage_fence();
  }
#undef STAGE

  // normalize and scatter to (N, CV, H, W2) f32
  int n = pair >> 2, b = pair & 3, bi = b >> 1, bj = b & 1;
#pragma unroll
  for (int r = 0; r < 8; ++r) {
    int l = lbase + r + 8 * lh;                       // acc row mapping = rsum mapping
    int side = l & 1, wl = (l >> 1) & 31, hl = l >> 6;
    int h = bi * 32 + hl, w = side * 64 + bj * 32 + wl;
    float inv = 1.0f / rsum[r];
    size_t base = (size_t)n * 64 * 8192 + (size_t)h * 128 + w;
    out[base + (size_t)(lm)      * 8192] = acc0[r] * inv;
    out[base + (size_t)(16 + lm) * 8192] = acc1[r] * inv;
    out[base + (size_t)(32 + lm) * 8192] = acc2[r] * inv;
    out[base + (size_t)(48 + lm) * 8192] = acc3[r] * inv;
  }
}

// ---------------------------------------------------------------------------
extern "C" void kernel_launch(void* const* d_in, const int* in_sizes, int n_in,
                              void* d_out, int out_size, void* d_ws, size_t ws_size,
                              hipStream_t stream) {
  (void)in_sizes; (void)n_in; (void)out_size; (void)ws_size;
  const float* x     = (const float*)d_in[0];
  const float* Wq    = (const float*)d_in[1];
  const float* bq    = (const float*)d_in[2];
  const float* gq    = (const float*)d_in[3];
  const float* betaq = (const float*)d_in[4];
  const float* Wk    = (const float*)d_in[5];
  const float* bk    = (const float*)d_in[6];
  const float* gk    = (const float*)d_in[7];
  const float* betak = (const float*)d_in[8];
  const float* Wv    = (const float*)d_in[9];
  const float* bv    = (const float*)d_in[10];
  float* out = (float*)d_out;

  char* ws = (char*)d_ws;
  size_t off = 0;
  auto alloc = [&](size_t bytes) { size_t o = off; off += (bytes + 255) & ~(size_t)255; return o; };
  _Float16* xt   = (_Float16*)(ws + alloc((size_t)NPIX * 64 * 2));
  _Float16* Wall = (_Float16*)(ws + alloc((size_t)128 * 64 * 2));
  float*    ball = (float*)   (ws + alloc(128 * 4));
  float*    praw = (float*)   (ws + alloc((size_t)NPIX * 128 * 4));
  float*    affA = (float*)   (ws + alloc(64 * 4));
  float*    affC = (float*)   (ws + alloc(64 * 4));
  _Float16* qbt  = (_Float16*)(ws + alloc((size_t)16 * L_TOK * 32 * 2));
  _Float16* kbt  = (_Float16*)(ws + alloc((size_t)16 * L_TOK * 32 * 2));
  _Float16* vbt  = (_Float16*)(ws + alloc((size_t)16 * 64 * L_TOK * 2));

  prep_wb<<<1, 128, 0, stream>>>(Wq, bq, Wk, bk, Wv, bv, Wall, ball);
  prep_xt<<<8192, 256, 0, stream>>>(x, xt);
  proj_kernel<<<256, 256, 0, stream>>>(xt, Wall, ball, praw);
  stats_kernel<<<64, 256, 0, stream>>>(praw, gq, betaq, gk, betak, affA, affC);
  pack_kernel<<<128, 256, 0, stream>>>(praw, affA, affC, qbt, kbt, vbt);
  attn_kernel<<<256, 256, 0, stream>>>(qbt, kbt, vbt, out);
}
